// STBlock_10471130268009
// MI455X (gfx1250) — matrix operand order repaired and checked by reading, not compiled, not run on hardware
//
#include <hip/hip_runtime.h>
#include <hip/hip_bf16.h>

// ---------------- constants ----------------
#define NS_   512
#define NI_   2048
#define NE_   2048
#define T_    96
#define D_    128
#define H_    4
#define C_    32
#define E_    4096
#define MS_   (NS_ * T_)   // 49152
#define MI_   (NI_ * T_)   // 196608
#define ME_   (NE_ * T_)   // 196608

typedef __attribute__((ext_vector_type(16))) _Float16 v16h;
typedef __attribute__((ext_vector_type(8)))  float    v8f;

__device__ __forceinline__ float sigmoidf_(float x) { return 1.0f / (1.0f + __expf(-x)); }

// ---------------- weight prep (transpose + cast to f16) ----------------
// conv weight (O,I,K=3) -> per-tap Bt[n=o][k=i] f16, 3 taps contiguous
__global__ void prep_conv_w(const float* __restrict__ w, _Float16* __restrict__ out) {
    int i = blockIdx.x * blockDim.x + threadIdx.x;
    if (i >= 3 * D_ * D_) return;
    int k = i / (D_ * D_);
    int r = i - k * D_ * D_;
    int o = r >> 7, ii = r & 127;
    out[i] = (_Float16)w[(o * D_ + ii) * 3 + k];
}
// GAT weight (K,N) -> Bt[n][k] f16
__global__ void prep_gat_w(const float* __restrict__ w, _Float16* __restrict__ out) {
    int i = blockIdx.x * blockDim.x + threadIdx.x;
    if (i >= D_ * D_) return;
    int n = i >> 7, k = i & 127;
    out[i] = (_Float16)w[k * D_ + n];
}

// ---- WMMA fragment helpers (contiguous k order, same for A and B) ----
// A (16x32): lanes 0-15 = rows; element e: K = half*16 + e  (contiguous 16)
__device__ __forceinline__ v16h load_a_frag(const float* __restrict__ arow, int kb, int half, float scale) {
    v16h a;
    int k0 = kb * 32 + half * 16;
#pragma unroll
    for (int e = 0; e < 8; ++e) {
        a[e]     = (_Float16)(arow[k0 + e] * scale);
        a[e + 8] = (_Float16)(arow[k0 + 8 + e] * scale);
    }
    return a;
}
// B (32x16): col = lane%16 ; element e: K = half*16 + e  (contiguous 16 halfs)
__device__ __forceinline__ v16h load_b_frag(const _Float16* __restrict__ brow, int kb, int half) {
    v16h b;
    int k0 = kb * 32 + half * 16;
#pragma unroll
    for (int e = 0; e < 16; ++e) b[e] = brow[k0 + e];
    return b;
}

// ---------------- WMMA GEMM: one wave computes a 16x128 strip ----------------
__global__ void gemm_wmma(const float* __restrict__ A, const _Float16* __restrict__ Bt,
                          const float* __restrict__ bias, float* __restrict__ C, int M) {
    int wid  = (blockIdx.x * blockDim.x + threadIdx.x) >> 5;
    int lane = threadIdx.x & 31;
    if (wid * 16 >= M) return;                      // wave-uniform
    int m0 = wid * 16;
    int half = lane >> 4;
    const float* arow = A + (size_t)(m0 + (lane & 15)) * D_;
    v16h afr[4];
#pragma unroll
    for (int kb = 0; kb < 4; ++kb) afr[kb] = load_a_frag(arow, kb, half, 1.0f);
#pragma unroll
    for (int nt = 0; nt < 8; ++nt) {
        const _Float16* brow = Bt + (size_t)(nt * 16 + (lane & 15)) * D_;
        v8f acc = {};
#pragma unroll
        for (int kb = 0; kb < 4; ++kb) {
            v16h b = load_b_frag(brow, kb, half);
            acc = __builtin_amdgcn_wmma_f32_16x16x32_f16(false, afr[kb], false, b, (short)0, acc, false, false);
        }
        int col = nt * 16 + (lane & 15);
        float bv = bias[col];
#pragma unroll
        for (int r = 0; r < 8; ++r)
            C[(size_t)(m0 + r + half * 8) * D_ + col] = acc[r] + bv;
    }
}

// ---------------- Conv1d(k=3,SAME) + bias + SiLU + residual; one wave = 16x128 strip ----------------
__global__ void conv_silu_res_wmma(const float* __restrict__ X, const _Float16* __restrict__ Wt,
                                   const float* __restrict__ bias, float* __restrict__ Y, int M) {
    int wid  = (blockIdx.x * blockDim.x + threadIdx.x) >> 5;
    int lane = threadIdx.x & 31;
    if (wid * 16 >= M) return;                      // wave-uniform
    int m0 = wid * 16;
    int half = lane >> 4;
    int row  = m0 + (lane & 15);
    int trow = (m0 % T_) + (lane & 15);             // T_ % 16 == 0 -> tile within one node
    v16h afr[3][4];
#pragma unroll
    for (int kk = 0; kk < 3; ++kk) {
        int   srct  = trow + kk - 1;
        float scale = (srct >= 0 && srct < T_) ? 1.0f : 0.0f;
        int   srow  = row + kk - 1;
        srow = srow < 0 ? 0 : (srow >= M ? M - 1 : srow);
        const float* arow = X + (size_t)srow * D_;
#pragma unroll
        for (int kb = 0; kb < 4; ++kb) afr[kk][kb] = load_a_frag(arow, kb, half, scale);
    }
#pragma unroll
    for (int nt = 0; nt < 8; ++nt) {
        v8f acc = {};
#pragma unroll
        for (int kk = 0; kk < 3; ++kk) {
            const _Float16* brow = Wt + ((size_t)kk * D_ + (nt * 16 + (lane & 15))) * D_;
#pragma unroll
            for (int kb = 0; kb < 4; ++kb) {
                v16h b = load_b_frag(brow, kb, half);
                acc = __builtin_amdgcn_wmma_f32_16x16x32_f16(false, afr[kk][kb], false, b, (short)0, acc, false, false);
            }
        }
        int col = nt * 16 + (lane & 15);
        float bv = bias[col];
#pragma unroll
        for (int r = 0; r < 8; ++r) {
            int orow = m0 + r + half * 8;
            float y = acc[r] + bv;
            y = y * sigmoidf_(y);
            Y[(size_t)orow * D_ + col] = y + X[(size_t)orow * D_ + col];
        }
    }
}

// ---------------- edge-attr projection: ep[r,d] = sum_j ea[r,j]*We[j,d] (K=8) ----------------
__global__ void edgeproj(const float* __restrict__ ea, const float* __restrict__ We,
                         float* __restrict__ ep, int E) {
    int i = blockIdx.x * blockDim.x + threadIdx.x;
    if (i >= E * D_) return;
    int r = i >> 7, d = i & 127;
    float s = 0.f;
#pragma unroll
    for (int j = 0; j < 8; ++j) s += ea[r * 8 + j] * We[j * D_ + d];
    ep[i] = s;
}

// ---------------- deterministic per-dst CSR over base edges ----------------
__global__ void csr_count(const int* __restrict__ ei, int* __restrict__ cnt, int E) {
    int j = blockIdx.x * blockDim.x + threadIdx.x;
    if (j >= E) return;
    atomicAdd(&cnt[ei[E + j]], 1);                  // integer atomics: order-independent
}
// one block of NS_ threads: inclusive LDS scan -> rowptr
__global__ void csr_scan(const int* __restrict__ cnt, int* __restrict__ rowptr) {
    __shared__ int sh[NS_];
    int tid = threadIdx.x;
    sh[tid] = cnt[tid];
    __syncthreads();
    for (int ofs = 1; ofs < NS_; ofs <<= 1) {
        int v = (tid >= ofs) ? sh[tid - ofs] : 0;
        __syncthreads();
        sh[tid] += v;
        __syncthreads();
    }
    rowptr[tid + 1] = sh[tid];
    if (tid == 0) rowptr[0] = 0;
}
// stable scatter: pos[j] = #{j' < j : dst[j'] == dst[j]}  (deterministic eidx content)
__global__ void csr_rank_scatter(const int* __restrict__ ei, const int* __restrict__ rowptr,
                                 int* __restrict__ eidx, int E) {
    int j = blockIdx.x * blockDim.x + threadIdx.x;
    if (j >= E) return;
    int d = ei[E + j];
    int pos = 0;
    for (int j2 = 0; j2 < j; ++j2) pos += (ei[E + j2] == d);
    eidx[rowptr[d] + pos] = j;
}

// ---------------- s2s self-loop mean attrs via CSR (deterministic) ----------------
__global__ void loop_mean_csr(const float* __restrict__ ea, const int* __restrict__ rowptr,
                              const int* __restrict__ eidx, float* __restrict__ mean) {
    int idx = blockIdx.x * blockDim.x + threadIdx.x;
    if (idx >= NS_ * 8) return;
    int d = idx >> 3, q = idx & 7;
    int beg = rowptr[d], end = rowptr[d + 1];
    float s = 0.f;
    for (int p = beg; p < end; ++p) s += ea[eidx[p] * 8 + q];
    mean[idx] = s / fmaxf((float)(end - beg), 1.0f);
}

// ---------------- GATv2 pass 1: logits (no atomics) ----------------
__global__ void gat_logits(const float* __restrict__ xl, const float* __restrict__ xr,
                           const float* __restrict__ ep, const int* __restrict__ ei,
                           const float* __restrict__ att, float* __restrict__ logit,
                           int E, int nloops, int nsrc, int ndst) {
    int i = blockIdx.x * blockDim.x + threadIdx.x;
    int nedge = E * T_;
    if (i >= nedge + nloops * T_) return;
    int src, dst, eprow;
    if (i < nedge) {
        int t = i / E, j = i - t * E;
        src = ei[j] + t * nsrc; dst = ei[E + j] + t * ndst; eprow = j;
    } else {
        int r = i - nedge;
        int t = r / nloops, d = r - t * nloops;
        src = dst = d + t * ndst; eprow = E + d;
    }
    const float* pl = xl + (size_t)src * D_;
    const float* pr = xr + (size_t)dst * D_;
    const float* pe = ep + (size_t)eprow * D_;
    float lg[H_] = {0.f, 0.f, 0.f, 0.f};
    for (int c = 0; c < D_; ++c) {
        float m = pl[c] + pr[c] + pe[c];
        m = (m > 0.f) ? m : 0.2f * m;
        lg[c >> 5] += m * att[c];
    }
#pragma unroll
    for (int h = 0; h < H_; ++h) logit[(size_t)i * H_ + h] = lg[h];
}

// ---------------- GATv2 pass 2: per-(dst,t,h) max + exp + den; ex overwrites logit ----------------
__global__ void gat_mxden(float* __restrict__ logit, const int* __restrict__ rowptr,
                          const int* __restrict__ eidx, float* __restrict__ den,
                          int E, int nloops) {
    int idx = blockIdx.x * blockDim.x + threadIdx.x;
    if (idx >= MS_ * H_) return;
    int h = idx & 3;
    int dstflat = idx >> 2;
    int d = dstflat % NS_, t = dstflat / NS_;
    int beg = rowptr[d], end = rowptr[d + 1];
    float mx = -3.0e38f;
    for (int p = beg; p < end; ++p) {
        float lg = logit[((size_t)t * E + eidx[p]) * H_ + h];
        mx = fmaxf(mx, lg);
    }
    size_t loopit = (size_t)E * T_ + (size_t)t * nloops + d;
    if (nloops) mx = fmaxf(mx, logit[loopit * H_ + h]);
    float s = 0.f;
    for (int p = beg; p < end; ++p) {
        size_t it = ((size_t)t * E + eidx[p]) * H_ + h;
        float ex = __expf(logit[it] - mx);
        logit[it] = ex;                 // each (edge-item, h) owned by exactly one dst thread
        s += ex;
    }
    if (nloops) {
        float ex = __expf(logit[loopit * H_ + h] - mx);
        logit[loopit * H_ + h] = ex;
        s += ex;
    }
    den[idx] = s;
}

// ---------------- GATv2 pass 3: gather-weighted sum + bias (no atomics) ----------------
__global__ void gat_out(const float* __restrict__ xl, const float* __restrict__ ex,
                        const float* __restrict__ den, const int* __restrict__ ei,
                        const int* __restrict__ rowptr, const int* __restrict__ eidx,
                        const float* __restrict__ bias, float* __restrict__ out,
                        int E, int nloops, int nsrc) {
    int idx = blockIdx.x * blockDim.x + threadIdx.x;
    if (idx >= MS_ * D_) return;
    int c = idx & 127;
    int dstflat = idx >> 7;
    int d = dstflat % NS_, t = dstflat / NS_;
    int h = c >> 5;
    int beg = rowptr[d], end = rowptr[d + 1];
    float s = 0.f;
    for (int p = beg; p < end; ++p) {
        int j = eidx[p];
        s += ex[((size_t)t * E + j) * H_ + h] * xl[((size_t)(ei[j] + t * nsrc)) * D_ + c];
    }
    if (nloops) {
        size_t loopit = (size_t)E * T_ + (size_t)t * nloops + d;
        s += ex[loopit * H_ + h] * xl[(size_t)dstflat * D_ + c];
    }
    float dn = den[(size_t)dstflat * H_ + h];
    out[idx] = (dn > 0.f ? s / dn : 0.f) + bias[c];
}

// ---------------- residual + LayerNorm ----------------
__global__ void residual_layernorm(const float* __restrict__ Hx, const float* __restrict__ R,
                                   const float* __restrict__ g, const float* __restrict__ b,
                                   float* __restrict__ out, int M) {
    int r = blockIdx.x * blockDim.x + threadIdx.x;
    if (r >= M) return;
    const float* ph = Hx + (size_t)r * D_;
    const float* pr = R  + (size_t)r * D_;
    float s = 0.f, s2 = 0.f;
    for (int c = 0; c < D_; ++c) { float v = ph[c] + pr[c]; s += v; s2 += v * v; }
    float mu  = s * (1.0f / D_);
    float var = s2 * (1.0f / D_) - mu * mu;
    float inv = rsqrtf(var + 1e-5f);
    float* po = out + (size_t)r * D_;
    for (int c = 0; c < D_; ++c) {
        float v = ph[c] + pr[c];
        po[c] = (v - mu) * inv * g[c] + b[c];
    }
}

// ---------------- host orchestration ----------------
static inline int cdiv(long a, long b) { return (int)((a + b - 1) / b); }

extern "C" void kernel_launch(void* const* d_in, const int* in_sizes, int n_in,
                              void* d_out, int out_size, void* d_ws, size_t ws_size,
                              hipStream_t stream) {
    // ---- inputs (setup_inputs dict insertion order; params flattened in insertion order) ----
    const float* h_s   = (const float*)d_in[0];
    const float* h_i   = (const float*)d_in[1];
    const float* h_e   = (const float*)d_in[2];
    const float* eaI   = (const float*)d_in[3];
    const float* eaE   = (const float*)d_in[4];
    const float* eaS   = (const float*)d_in[5];
    const int*   eiI   = (const int*)d_in[6];
    const int*   eiE   = (const int*)d_in[7];
    const int*   eiS   = (const int*)d_in[8];
    const float* cw_s  = (const float*)d_in[9];
    const float* cb_s  = (const float*)d_in[10];
    const float* cw_i  = (const float*)d_in[11];
    const float* cb_i  = (const float*)d_in[12];
    const float* cw_e  = (const float*)d_in[13];
    const float* cb_e  = (const float*)d_in[14];
    // g_i2s = 15..21 (Wl,bl,Wr,br,We,att,bias), g_e2s = 22..28, g_s2s = 29..35
    const float* ln_g  = (const float*)d_in[36];
    const float* ln_b  = (const float*)d_in[37];

    // ---- workspace layout ----
    char* ws = (char*)d_ws;
    size_t off = 0;
    auto take = [&](size_t bytes) { void* p = ws + off; off = (off + bytes + 255) & ~(size_t)255; return p; };
    _Float16* wF16   = (_Float16*)take((size_t)(9 + 6) * D_ * D_ * 2);
    float* hs_conv   = (float*)take((size_t)MS_ * D_ * 4);
    float* xl        = (float*)take((size_t)MI_ * D_ * 4);
    float* xr        = (float*)take((size_t)MS_ * D_ * 4);
    float* ep        = (float*)take((size_t)(E_ + NS_) * D_ * 4);
    float* logit     = (float*)take((size_t)(E_ * T_ + NS_ * T_) * H_ * 4);
    float* den       = (float*)take((size_t)MS_ * H_ * 4);
    float* hbuf      = (float*)take((size_t)MS_ * D_ * 4);
    float* lsum      = (float*)take((size_t)NS_ * 8 * 4);
    int*   csr_cnt   = (int*)take((size_t)NS_ * 4);
    int*   csr_rowp  = (int*)take((size_t)(NS_ + 1) * 4);
    int*   csr_eidx  = (int*)take((size_t)E_ * 4);

    _Float16* wConvS = wF16;
    _Float16* wConvI = wF16 + 3 * D_ * D_;
    _Float16* wConvE = wF16 + 6 * D_ * D_;
    _Float16* wGat   = wF16 + 9 * D_ * D_;   // [g*2+0]=WlT, [g*2+1]=WrT

    float* outHs = (float*)d_out;
    float* outHi = (float*)d_out + (size_t)MS_ * D_;
    float* outHe = (float*)d_out + (size_t)(MS_ + MI_) * D_;

    // ---- weight prep ----
    prep_conv_w<<<cdiv(3 * D_ * D_, 256), 256, 0, stream>>>(cw_s, wConvS);
    prep_conv_w<<<cdiv(3 * D_ * D_, 256), 256, 0, stream>>>(cw_i, wConvI);
    prep_conv_w<<<cdiv(3 * D_ * D_, 256), 256, 0, stream>>>(cw_e, wConvE);
    for (int g = 0; g < 3; ++g) {
        const float* Wl = (const float*)d_in[15 + g * 7 + 0];
        const float* Wr = (const float*)d_in[15 + g * 7 + 2];
        prep_gat_w<<<cdiv(D_ * D_, 256), 256, 0, stream>>>(Wl, wGat + (size_t)(g * 2 + 0) * D_ * D_);
        prep_gat_w<<<cdiv(D_ * D_, 256), 256, 0, stream>>>(Wr, wGat + (size_t)(g * 2 + 1) * D_ * D_);
    }

    // ---- conv + SiLU + residual (WMMA). h_i/h_e go straight to d_out. ----
    conv_silu_res_wmma<<<cdiv((size_t)(MS_ / 16) * 32, 256), 256, 0, stream>>>(h_s, wConvS, cb_s, hs_conv, MS_);
    conv_silu_res_wmma<<<cdiv((size_t)(MI_ / 16) * 32, 256), 256, 0, stream>>>(h_i, wConvI, cb_i, outHi, MI_);
    conv_silu_res_wmma<<<cdiv((size_t)(ME_ / 16) * 32, 256), 256, 0, stream>>>(h_e, wConvE, cb_e, outHe, ME_);

    // helper to run one GATv2 layer
    auto run_gat = [&](const float* x_src, int Msrc, int nsrc, const float* x_dst,
                       const int* ei, const float* ea, int g, int nloops, float* h_out) {
        const float* bl   = (const float*)d_in[15 + g * 7 + 1];
        const float* br   = (const float*)d_in[15 + g * 7 + 3];
        const float* We   = (const float*)d_in[15 + g * 7 + 4];
        const float* att  = (const float*)d_in[15 + g * 7 + 5];
        const float* bias = (const float*)d_in[15 + g * 7 + 6];
        // deterministic CSR over base edges
        hipMemsetAsync(csr_cnt, 0, (size_t)NS_ * 4, stream);
        csr_count<<<cdiv(E_, 256), 256, 0, stream>>>(ei, csr_cnt, E_);
        csr_scan<<<1, NS_, 0, stream>>>(csr_cnt, csr_rowp);
        csr_rank_scatter<<<cdiv(E_, 256), 256, 0, stream>>>(ei, csr_rowp, csr_eidx, E_);
        // projections
        gemm_wmma<<<cdiv((size_t)(Msrc / 16) * 32, 256), 256, 0, stream>>>(
            x_src, wGat + (size_t)(g * 2 + 0) * D_ * D_, bl, xl, Msrc);
        gemm_wmma<<<cdiv((size_t)(MS_ / 16) * 32, 256), 256, 0, stream>>>(
            x_dst, wGat + (size_t)(g * 2 + 1) * D_ * D_, br, xr, MS_);
        edgeproj<<<cdiv(E_ * D_, 256), 256, 0, stream>>>(ea, We, ep, E_);
        if (nloops > 0) {
            loop_mean_csr<<<cdiv(NS_ * 8, 256), 256, 0, stream>>>(ea, csr_rowp, csr_eidx, lsum);
            edgeproj<<<cdiv(nloops * D_, 256), 256, 0, stream>>>(lsum, We, ep + (size_t)E_ * D_, nloops);
        }
        int nitems = E_ * T_ + nloops * T_;
        gat_logits<<<cdiv(nitems, 256), 256, 0, stream>>>(xl, xr, ep, ei, att, logit, E_, nloops, nsrc, NS_);
        gat_mxden<<<cdiv(MS_ * H_, 256), 256, 0, stream>>>(logit, csr_rowp, csr_eidx, den, E_, nloops);
        gat_out<<<cdiv(MS_ * D_, 256), 256, 0, stream>>>(xl, logit, den, ei, csr_rowp, csr_eidx,
                                                         bias, h_out, E_, nloops, nsrc);
    };

    // ---- GAT i2s: src=h_i(conv), dst=hs_conv -> h1 (hbuf) ----
    run_gat(outHi, MI_, NI_, hs_conv, eiI, eaI, 0, 0, hbuf);
    // ---- GAT e2s: src=h_e(conv), dst=h1 -> h2 (hbuf) ----
    run_gat(outHe, ME_, NE_, hbuf, eiE, eaE, 1, 0, hbuf);
    // ---- GAT s2s with mean self-loops: src=dst=h2 -> h3 (hbuf) ----
    run_gat(hbuf, MS_, NS_, hbuf, eiS, eaS, 2, NS_, hbuf);

    // ---- residual + LayerNorm -> d_out[0] ----
    residual_layernorm<<<cdiv(MS_, 256), 256, 0, stream>>>(hbuf, hs_conv, ln_g, ln_b, outHs, MS_);
    (void)in_sizes; (void)n_in; (void)out_size; (void)ws_size;
}
